// MultiHeadAttention_6992206757930
// MI455X (gfx1250) — compile-verified, matrix operand order checked by reading, not run on hardware
//
#include <hip/hip_runtime.h>
#include <hip/hip_bf16.h>
#include <stdint.h>

#define BATCH   2
#define SEQ     2048
#define HIDDEN  1024
#define EMBED   1024
#define HEADS   16
#define HD      64

typedef __attribute__((ext_vector_type(16))) __bf16 v16bf;
typedef __attribute__((ext_vector_type(8)))  float  v8f;

union FragAB { v16bf v; uint32_t u[8]; };

#if defined(__HIP_DEVICE_COMPILE__) && \
    __has_builtin(__builtin_amdgcn_global_load_async_to_lds_b128) && \
    __has_builtin(__builtin_amdgcn_s_wait_asynccnt)
#define USE_ASYNC_LDS 1
#else
#define USE_ASYNC_LDS 0
#endif

#if USE_ASYNC_LDS
// Builtin signature (from hipcc diagnostic): pointee is a 16-byte int vector.
typedef int v4i_b128 __attribute__((vector_size(4 * sizeof(int))));
typedef __attribute__((address_space(1))) v4i_b128* gbl_v4i_p;
typedef __attribute__((address_space(3))) v4i_b128* lds_v4i_p;
#endif

__device__ __forceinline__ uint16_t f2bf(float f) {
    // round-to-nearest-even fp32 -> bf16
    uint32_t x = __float_as_uint(f);
    return (uint16_t)((x + 0x7FFFu + ((x >> 16) & 1u)) >> 16);
}

__device__ __forceinline__ v8f wmma_bf16(const FragAB& a, const FragAB& b, v8f c) {
    return __builtin_amdgcn_wmma_f32_16x16x32_bf16(false, a.v, false, b.v,
                                                   (short)0, c, false, false);
}

// 16-byte global -> LDS copy; async (ASYNCcnt) on gfx1250 toolchains that
// declare the builtin, synchronous VGPR round-trip otherwise.
__device__ __forceinline__ void cp16_g2l(const uint16_t* g, uint16_t* l) {
#if USE_ASYNC_LDS
    __builtin_amdgcn_global_load_async_to_lds_b128(
        (gbl_v4i_p)(void*)g, (lds_v4i_p)(void*)l, 0, 0);
#else
    *(uint4*)l = *(const uint4*)g;
#endif
}

__device__ __forceinline__ void wait_cp_async() {
#if USE_ASYNC_LDS
    __builtin_amdgcn_s_wait_asynccnt(0);
#endif
}

// ---------------------------------------------------------------------------
// Kernel 1: Y = X*W + bias, fp32 in, bf16 out.
// blockIdx.z: 0 -> Q ([B,H,S,D]), 1 -> K ([B,H,S,D]), 2 -> V ([B,H,D,S], V^T)
// Block tile 128(M) x 128(N), 8 waves as 4(m) x 2(n): each wave owns a
// 32x64 sub-tile = 2 A-frags x 4 B-frags -> 8 WMMAs with 2x fragment reuse.
// ---------------------------------------------------------------------------
__global__ __launch_bounds__(256) void qkv_proj_kernel(
    const float* __restrict__ xq, const float* __restrict__ xk, const float* __restrict__ xv,
    const float* __restrict__ wq, const float* __restrict__ bq,
    const float* __restrict__ wk, const float* __restrict__ bk,
    const float* __restrict__ wv, const float* __restrict__ bv,
    uint16_t* __restrict__ qws, uint16_t* __restrict__ kws, uint16_t* __restrict__ vws)
{
    const int mode = blockIdx.z;
    const float* __restrict__ X    = (mode == 0) ? xq : (mode == 1) ? xk : xv;
    const float* __restrict__ W    = (mode == 0) ? wq : (mode == 1) ? wk : wv;
    const float* __restrict__ bias = (mode == 0) ? bq : (mode == 1) ? bk : bv;

    __shared__ uint16_t sA[128 * 32];   // [m][k] row-major, bf16 bits
    __shared__ uint16_t sB[128 * 32];   // [n][k] n-major, bf16 bits

    const int tid  = threadIdx.x;
    const int wave = tid >> 5;
    const int lane = tid & 31;
    const int half = lane >> 4;
    const int l16  = lane & 15;

    const int mgrp = wave >> 1;         // 0..3 -> 32-row group
    const int ngrp = wave & 1;          // 0..1 -> 64-col group

    const int Mbase = blockIdx.y * 128;
    const int Nbase = blockIdx.x * 128;

    // Per-thread staging coordinates.
    const int mr = tid >> 1;            // A: row, 2 threads/row
    const int kkA = (tid & 1) * 16;     // A: 16 floats each
    const int kr = tid >> 3;            // B: k-row, 8 threads/row
    const int nn = (tid & 7) * 16;      // B: 16 n's each

    const v8f vzero = {0.f, 0.f, 0.f, 0.f, 0.f, 0.f, 0.f, 0.f};
    v8f acc[2][4];
#pragma unroll
    for (int i = 0; i < 2; i++)
#pragma unroll
        for (int j = 0; j < 4; j++) acc[i][j] = vzero;

    for (int kc = 0; kc < HIDDEN; kc += 32) {
        if (kc + 32 < HIDDEN) {
            __builtin_prefetch(X + (size_t)(Mbase + mr) * HIDDEN + kc + 32 + kkA, 0, 1);
            __builtin_prefetch(W + (size_t)(kc + 32 + kr) * EMBED + Nbase + nn, 0, 1);
        }
        // Stage X tile (fp32 -> bf16).
        {
            const float* src = X + (size_t)(Mbase + mr) * HIDDEN + kc + kkA;
            uint32_t* dst = (uint32_t*)&sA[mr * 32 + kkA];
#pragma unroll
            for (int i = 0; i < 8; i++) {
                float f0 = src[2 * i], f1 = src[2 * i + 1];
                dst[i] = (uint32_t)f2bf(f0) | ((uint32_t)f2bf(f1) << 16);
            }
        }
        // Stage W tile transposed to n-major.
        {
            const float* src = W + (size_t)(kc + kr) * EMBED + Nbase + nn;
#pragma unroll
            for (int i = 0; i < 16; i++)
                sB[(nn + i) * 32 + kr] = f2bf(src[i]);
        }
        __syncthreads();

        // Preload 2 A-fragments + 4 B-fragments, then 8 WMMAs (2x reuse each).
        FragAB af[2];
#pragma unroll
        for (int i = 0; i < 2; i++) {
            const uint16_t* p = &sA[(mgrp * 32 + i * 16 + l16) * 32 + half * 8];
            *(uint4*)&af[i].u[0] = *(const uint4*)p;
            *(uint4*)&af[i].u[4] = *(const uint4*)(p + 16);
        }
        FragAB bfrag[4];
#pragma unroll
        for (int j = 0; j < 4; j++) {
            const uint16_t* p = &sB[(ngrp * 64 + j * 16 + l16) * 32 + half * 8];
            *(uint4*)&bfrag[j].u[0] = *(const uint4*)p;
            *(uint4*)&bfrag[j].u[4] = *(const uint4*)(p + 16);
        }
#pragma unroll
        for (int i = 0; i < 2; i++)
#pragma unroll
            for (int j = 0; j < 4; j++)
                acc[i][j] = wmma_bf16(af[i], bfrag[j], acc[i][j]);
        __syncthreads();
    }

    // Epilogue: add bias, cast bf16, scatter to head-split workspace layouts.
#pragma unroll
    for (int i = 0; i < 2; i++) {
#pragma unroll
        for (int j = 0; j < 4; j++) {
            const int col = Nbase + ngrp * 64 + j * 16 + l16;
            const int h = col >> 6, d = col & 63;
            const float bb = bias[col];
#pragma unroll
            for (int r = 0; r < 8; r++) {
                const int row = Mbase + mgrp * 32 + i * 16 + r + 8 * half;
                const int b = row >> 11, s = row & (SEQ - 1);
                const uint16_t val = f2bf(acc[i][j][r] + bb);
                if (mode == 2)
                    vws[((size_t)(b * HEADS + h) * HD + d) * SEQ + s] = val;  // V^T
                else if (mode == 0)
                    qws[((size_t)(b * HEADS + h) * SEQ + s) * HD + d] = val;
                else
                    kws[((size_t)(b * HEADS + h) * SEQ + s) * HD + d] = val;
            }
        }
    }
}

// ---------------------------------------------------------------------------
// Kernel 2: fused flash attention. One block = (b, h, 128 q rows); 8 waves,
// each wave owns 16 q rows. Keys in chunks of 64 with online softmax.
// K/V tiles double-buffered in LDS, filled by async global->LDS copies that
// overlap the previous tile's WMMAs.
// ---------------------------------------------------------------------------
__global__ __launch_bounds__(256) void flash_attn_kernel(
    const uint16_t* __restrict__ qws, const uint16_t* __restrict__ kws,
    const uint16_t* __restrict__ vws, float* __restrict__ out)
{
    __shared__ uint16_t sK[2][64 * 64];       // [key][d]
    __shared__ uint16_t sV[2][64 * 64];       // [d][key]  (V^T)
    __shared__ uint16_t sP[8 * 16 * 64];      // per-wave [q][key] prob tile

    const int tid  = threadIdx.x;
    const int wave = tid >> 5;
    const int lane = tid & 31;
    const int half = lane >> 4;
    const int l16  = lane & 15;

    const int h = blockIdx.y;
    const int b = blockIdx.z;
    const int qbase = blockIdx.x * 128 + wave * 16;

    const size_t headSD = (size_t)(b * HEADS + h) * SEQ * HD;  // Q/K base
    const size_t headDS = (size_t)(b * HEADS + h) * HD * SEQ;  // V^T base

    // Per-thread staging coordinates: 256 threads x 2 x 16B per 8KB tile.
    const int srow = tid >> 2;                // key (for K) / d (for V)
    const int scol = (tid & 3) * 16;
    const uint16_t* gK = kws + headSD + (size_t)srow * HD + scol;   // + k0*HD
    const uint16_t* gV = vws + headDS + (size_t)srow * SEQ + scol;  // + k0
    const int loff = srow * 64 + scol;

    // Q fragments for this wave's 16 rows (two 32-wide chunks over d).
    FragAB qf[2];
#pragma unroll
    for (int c = 0; c < 2; c++) {
        const uint16_t* p = qws + headSD + (size_t)(qbase + l16) * HD + c * 32 + half * 8;
        *(uint4*)&qf[c].u[0] = *(const uint4*)p;
        *(uint4*)&qf[c].u[4] = *(const uint4*)(p + 16);
    }

    const v8f vzero = {0.f, 0.f, 0.f, 0.f, 0.f, 0.f, 0.f, 0.f};
    v8f acc[4];
    float mrow[8], lrow[8];
#pragma unroll
    for (int t = 0; t < 4; t++) acc[t] = vzero;
#pragma unroll
    for (int r = 0; r < 8; r++) { mrow[r] = -1e30f; lrow[r] = 0.f; }

    const float scale = 0.125f;  // 1/sqrt(64)

    // Prologue: stage tile 0 into buffer 0.
    cp16_g2l(gK, &sK[0][loff]);
    cp16_g2l(gK + 8, &sK[0][loff + 8]);
    cp16_g2l(gV, &sV[0][loff]);
    cp16_g2l(gV + 8, &sV[0][loff + 8]);

    for (int it = 0; it < SEQ / 64; ++it) {
        const int p = it & 1;
        wait_cp_async();       // this wave's copies into buffer p are done
        __syncthreads();       // everyone's copies done + prev compute on p^1 done

        if (it + 1 < SEQ / 64) {   // stage next tile into the other buffer
            const int k0n = (it + 1) * 64;
            cp16_g2l(gK + (size_t)k0n * HD, &sK[p ^ 1][loff]);
            cp16_g2l(gK + (size_t)k0n * HD + 8, &sK[p ^ 1][loff + 8]);
            cp16_g2l(gV + k0n, &sV[p ^ 1][loff]);
            cp16_g2l(gV + k0n + 8, &sV[p ^ 1][loff + 8]);
        }

        const uint16_t* K0 = sK[p];
        const uint16_t* V0 = sV[p];

        // Scores: preload all 8 K fragments, then 8 chained WMMAs.
        FragAB kf[8];
#pragma unroll
        for (int t = 0; t < 4; t++) {
            const uint16_t* q0 = &K0[(t * 16 + l16) * 64 + half * 8];
            *(uint4*)&kf[2 * t].u[0]     = *(const uint4*)q0;
            *(uint4*)&kf[2 * t].u[4]     = *(const uint4*)(q0 + 16);
            *(uint4*)&kf[2 * t + 1].u[0] = *(const uint4*)(q0 + 32);
            *(uint4*)&kf[2 * t + 1].u[4] = *(const uint4*)(q0 + 48);
        }
        v8f sc[4];
#pragma unroll
        for (int t = 0; t < 4; t++) {
            v8f z = wmma_bf16(qf[0], kf[2 * t], vzero);
            sc[t] = wmma_bf16(qf[1], kf[2 * t + 1], z);
        }
#pragma unroll
        for (int t = 0; t < 4; t++) sc[t] *= scale;

        // Online softmax, per row r (row = qbase + r + 8*half).
#pragma unroll
        for (int r = 0; r < 8; r++) {
            float mx = fmaxf(fmaxf(sc[0][r], sc[1][r]), fmaxf(sc[2][r], sc[3][r]));
#pragma unroll
            for (int o = 1; o <= 8; o <<= 1) mx = fmaxf(mx, __shfl_xor(mx, o, 32));
            const float mn = fmaxf(mrow[r], mx);
            const float corr = __expf(mrow[r] - mn);
            mrow[r] = mn;
            lrow[r] *= corr;
#pragma unroll
            for (int t = 0; t < 4; t++) acc[t][r] *= corr;
            float rs = 0.f;
#pragma unroll
            for (int t = 0; t < 4; t++) {
                const float pe = __expf(sc[t][r] - mn);
                sc[t][r] = pe;
                rs += pe;
            }
#pragma unroll
            for (int o = 1; o <= 8; o <<= 1) rs += __shfl_xor(rs, o, 32);
            lrow[r] += rs;
        }

        // P: C-layout -> per-wave LDS [q][key] bf16 -> A-operand fragments.
        uint16_t* myP = &sP[wave * 16 * 64];
#pragma unroll
        for (int t = 0; t < 4; t++)
#pragma unroll
            for (int r = 0; r < 8; r++)
                myP[(r + 8 * half) * 64 + t * 16 + l16] = f2bf(sc[t][r]);

        FragAB pf[2];
        {
            const uint16_t* q0 = myP + l16 * 64 + half * 8;
            *(uint4*)&pf[0].u[0] = *(const uint4*)q0;
            *(uint4*)&pf[0].u[4] = *(const uint4*)(q0 + 16);
            *(uint4*)&pf[1].u[0] = *(const uint4*)(q0 + 32);
            *(uint4*)&pf[1].u[4] = *(const uint4*)(q0 + 48);
        }

        // O += P * V: preload all 8 V fragments, then 8 chained WMMAs.
        FragAB vf[8];
#pragma unroll
        for (int t = 0; t < 4; t++) {
            const uint16_t* q0 = &V0[(t * 16 + l16) * 64 + half * 8];
            *(uint4*)&vf[2 * t].u[0]     = *(const uint4*)q0;
            *(uint4*)&vf[2 * t].u[4]     = *(const uint4*)(q0 + 16);
            *(uint4*)&vf[2 * t + 1].u[0] = *(const uint4*)(q0 + 32);
            *(uint4*)&vf[2 * t + 1].u[4] = *(const uint4*)(q0 + 48);
        }
#pragma unroll
        for (int t = 0; t < 4; t++) {
            acc[t] = wmma_bf16(pf[0], vf[2 * t], acc[t]);
            acc[t] = wmma_bf16(pf[1], vf[2 * t + 1], acc[t]);
        }
        // No trailing barrier: the top-of-loop barrier orders the next
        // buffer overwrite against this iteration's readers.
    }

    // Epilogue: normalize and write fp32 [B,S,E] with E = h*64 + d.
#pragma unroll
    for (int t = 0; t < 4; t++) {
        const int d = t * 16 + l16;
#pragma unroll
        for (int r = 0; r < 8; r++) {
            const int row = qbase + r + 8 * half;
            out[(size_t)(b * SEQ + row) * EMBED + h * HD + d] = acc[t][r] / lrow[r];
        }
    }
}

// ---------------------------------------------------------------------------
extern "C" void kernel_launch(void* const* d_in, const int* in_sizes, int n_in,
                              void* d_out, int out_size, void* d_ws, size_t ws_size,
                              hipStream_t stream) {
    (void)in_sizes; (void)n_in; (void)out_size; (void)ws_size;
    const float* q  = (const float*)d_in[0];
    const float* k  = (const float*)d_in[1];
    const float* v  = (const float*)d_in[2];
    const float* wq = (const float*)d_in[3];
    const float* bq = (const float*)d_in[4];
    const float* wk = (const float*)d_in[5];
    const float* bk = (const float*)d_in[6];
    const float* wv = (const float*)d_in[7];
    const float* bv = (const float*)d_in[8];
    float* out = (float*)d_out;

    // Workspace: 3 x [B,H,S,D] bf16 = 3 x 8 MB = 24 MB.
    const size_t hsd = (size_t)BATCH * HEADS * SEQ * HD;
    uint16_t* qws = (uint16_t*)d_ws;
    uint16_t* kws = qws + hsd;
    uint16_t* vws = kws + hsd;

    dim3 gGemm(EMBED / 128, (BATCH * SEQ) / 128, 3);
    qkv_proj_kernel<<<gGemm, 256, 0, stream>>>(q, k, v, wq, bq, wk, bk, wv, bv,
                                               qws, kws, vws);

    dim3 gAttn(SEQ / 128, HEADS, BATCH);
    flash_attn_kernel<<<gAttn, 256, 0, stream>>>(qws, kws, vws, out);
}
